// Attention_C_12111807775306
// MI455X (gfx1250) — matrix operand order repaired and checked by reading, not compiled, not run on hardware
//
#include <hip/hip_runtime.h>
#include <stdint.h>

// ---------------------------------------------------------------------------
// MI455X (gfx1250, wave32) implementation.
// Conv/GEMM work runs on v_wmma_f32_16x16x32_bf16 with block-cooperative LDS
// staging: one block = all M-tiles of an output strip, so activation (B) tiles
// are fetched from L2 once per block instead of once per M-tile (12-24x less
// L2 traffic on the dominant ex_out conv). Reductions/softmax/depthwise on VALU.
// Stage-1 (attention_ca + ca1_proj) computed for batch 0 only: the reference
// uses it solely for the batch-0 expert gate.
// ---------------------------------------------------------------------------

typedef __attribute__((ext_vector_type(16))) __bf16 v16bf;
typedef __attribute__((ext_vector_type(8)))  float  v8f;

#define DEV __device__ __forceinline__

DEV unsigned short f2bf(float f) {
  union { float f; unsigned u; } v; v.f = f;
  unsigned r = v.u + 0x7FFFu + ((v.u >> 16) & 1u);   // RNE
  return (unsigned short)(r >> 16);
}
DEV float bf2f(unsigned short h) {
  union { unsigned u; float f; } v; v.u = ((unsigned)h) << 16; return v.f;
}

DEV v8f wmma_bf16(v16bf a, v16bf b, v8f c) {
  return __builtin_amdgcn_wmma_f32_16x16x32_bf16(false, a, false, b, (short)0, c,
                                                 false, false);
}

// A fragment, 16x32 bf16 (ISA layout): lane holds row m = lane&15; K-half kh = lane>>4.
// elements 0..7 -> K = kh*8+e ; elements 8..15 -> K = 16+kh*8+(e-8)
DEV v16bf frag_a(const unsigned short* arow, int kh) {
  v16bf a;
  __builtin_memcpy(&a, arow + kh * 8, 16);
  __builtin_memcpy((char*)&a + 16, arow + 16 + kh * 8, 16);
  return a;
}
// B fragment, 32x16 bf16: lane holds column n = lane&15; K order as frag_a (row stride 64).
DEV v16bf frag_b(const unsigned short* p, int kh) {
  v16bf b;
  for (int s = 0; s < 16; ++s) {
    int k = (s < 8) ? kh * 8 + s : 16 + kh * 8 + (s - 8);
    __builtin_memcpy((char*)&b + 2 * s, p + k * 64, 2);
  }
  return b;
}

// ---------------------------------------------------------------------------
// Utility kernels
// ---------------------------------------------------------------------------
__global__ void k_zero32(uint32_t* p, size_t n) {
  size_t i = (size_t)blockIdx.x * blockDim.x + threadIdx.x;
  size_t st = (size_t)gridDim.x * blockDim.x;
  for (; i < n; i += st) p[i] = 0u;
}

__global__ void k_f2bf(const float* __restrict__ in, unsigned short* __restrict__ out, int n) {
  int i = blockIdx.x * blockDim.x + threadIdx.x;
  int st = gridDim.x * blockDim.x;
  for (; i < n; i += st) out[i] = f2bf(in[i]);
}

// [M,K,ks,ks] f32 -> tap-major [ks*ks][M][K] bf16 (contiguous WMMA A rows)
__global__ void k_repack(const float* __restrict__ w, unsigned short* __restrict__ out,
                         int M, int K, int ks) {
  int t2 = ks * ks; int n = M * K * t2;
  int i = blockIdx.x * blockDim.x + threadIdx.x;
  int st = gridDim.x * blockDim.x;
  for (; i < n; i += st) {
    int o = i / (K * t2); int rem = i - o * (K * t2);
    int c = rem / t2;     int tap = rem - c * t2;
    out[((size_t)tap * M + o) * K + c] = f2bf(w[i]);
  }
}

// I f32 [4,192,64,64] -> bf16 padded [4,192,68,68] interior (halo pre-zeroed)
__global__ void k_pad_in(const float* __restrict__ I, unsigned short* __restrict__ xpad) {
  int i = blockIdx.x * blockDim.x + threadIdx.x;
  if (i >= 4 * 192 * 4096) return;
  int l = i & 4095; int bc = i >> 12;
  int y = l >> 6, x = l & 63;
  xpad[((size_t)bc * 68 + (y + 2)) * 68 + (x + 2)] = f2bf(I[i]);
}

// ---------------------------------------------------------------------------
// Stage 1: pooled-kv rows, l2 norms, channel attention (batch 0 only)
// ---------------------------------------------------------------------------
__global__ __launch_bounds__(256) void k_ca_pool_norm(const float* __restrict__ T,
                                                      unsigned short* __restrict__ kn) {
  __shared__ float rowv[4096];
  __shared__ float red[256];
  int cp = blockIdx.x, t = threadIdx.x;              // cp in [0,48)
  const float* t0 = T + (size_t)(4 * cp) * 4096;     // batch 0
  float ss = 0.f;
  for (int l = t; l < 4096; l += 256) {
    float v = 0.25f * (t0[l] + t0[4096 + l] + t0[8192 + l] + t0[12288 + l]);
    rowv[l] = v; ss += v * v;
  }
  red[t] = ss; __syncthreads();
  for (int s = 128; s > 0; s >>= 1) { if (t < s) red[t] += red[t + s]; __syncthreads(); }
  float inv = 1.f / fmaxf(sqrtf(red[0]), 1e-12f);
  for (int l = t; l < 4096; l += 256) kn[(size_t)cp * 4096 + l] = f2bf(rowv[l] * inv);
}

// 1/||row||_2 ; row = b*nc + c ; addr = X + (b*CB + c0 + c)*L
__global__ __launch_bounds__(256) void k_row_inv_norm(const float* __restrict__ X,
                                                      float* __restrict__ inv,
                                                      int nc, int CB, int c0, int L) {
  __shared__ float red[256];
  int row = blockIdx.x;
  int b = row / nc, c = row - b * nc;
  const float* p = X + ((size_t)b * CB + c0 + c) * (size_t)L;
  float ss = 0.f;
  for (int l = threadIdx.x; l < L; l += 256) { float v = p[l]; ss += v * v; }
  red[threadIdx.x] = ss; __syncthreads();
  for (int s = 128; s > 0; s >>= 1) { if (threadIdx.x < s) red[threadIdx.x] += red[threadIdx.x + s]; __syncthreads(); }
  if (threadIdx.x == 0) inv[row] = 1.f / fmaxf(sqrtf(red[0]), 1e-12f);
}

// Channel attention (b=0): 4x channel tiling collapses to softmax over 8 rows.
__global__ __launch_bounds__(256) void k_ca_attn(const float* __restrict__ I,
                                                 const unsigned short* __restrict__ kn,
                                                 const float* __restrict__ qinv,
                                                 unsigned short* __restrict__ outca) {
  __shared__ float qt[32][64];
  __shared__ float kt8[8][64];
  __shared__ float A[32][8];
  __shared__ float kt[8][256];
  int h = blockIdx.x, t = threadIdx.x;
  const float* qb = I + (size_t)h * 32 * 4096;            // batch 0
  const unsigned short* knb = kn + (size_t)h * 8 * 4096;
  int c = t >> 3, e = t & 7;
  float g = 0.f;
  for (int l0 = 0; l0 < 4096; l0 += 64) {
    for (int i = t; i < 32 * 64; i += 256) { int cc = i >> 6, ll = i & 63; qt[cc][ll] = qb[(size_t)cc * 4096 + l0 + ll]; }
    for (int i = t; i < 8 * 64;  i += 256) { int ee = i >> 6, ll = i & 63; kt8[ee][ll] = bf2f(knb[(size_t)ee * 4096 + l0 + ll]); }
    __syncthreads();
#pragma unroll 4
    for (int j = 0; j < 64; ++j) g += qt[c][j] * kt8[e][j];
    __syncthreads();
  }
  A[c][e] = g * qinv[h * 32 + c];
  __syncthreads();
  if (t < 32) {
    float mx = A[t][0];
    for (int d = 1; d < 8; ++d) mx = fmaxf(mx, A[t][d]);
    float sum = 0.f;
    for (int d = 0; d < 8; ++d) { float ev = __expf(A[t][d] - mx); A[t][d] = ev; sum += ev; }
    float is = 1.f / sum;
    for (int d = 0; d < 8; ++d) A[t][d] *= is;
  }
  __syncthreads();
  for (int l0 = 0; l0 < 4096; l0 += 256) {
    for (int i = t; i < 8 * 256; i += 256) { int ee = i >> 8, ll = i & 255; kt[ee][ll] = bf2f(knb[(size_t)ee * 4096 + l0 + ll]); }
    __syncthreads();
    for (int cc = 0; cc < 32; ++cc) {
      float a0 = 0.f;
#pragma unroll
      for (int d = 0; d < 8; ++d) a0 += A[cc][d] * kt[d][t];
      outca[((size_t)h * 32 + cc) * 4096 + l0 + t] = f2bf(a0);
    }
    __syncthreads();
  }
}

// ---------------------------------------------------------------------------
// Cooperative bf16 WMMA GEMM (1x1 conv): block = ALL M-tiles of one (b, 64-col
// N strip); blockDim = (M/16)*32. A tile [M][32] and B tile [32][64] are staged
// in LDS once per K-chunk and shared by every wave -> B read from L2 once.
// ---------------------------------------------------------------------------
__global__ __launch_bounds__(768) void k_gemm_coop(const unsigned short* __restrict__ W,
                                                   const unsigned short* __restrict__ X,
                                                   float* __restrict__ Yf,
                                                   unsigned short* __restrict__ Ybf,
                                                   const float* __restrict__ bias,
                                                   int M, int K, int L, int batch) {
  extern __shared__ unsigned short sh[];
  unsigned short* shA = sh;                      // [M][32]
  unsigned short* shB = sh + (size_t)M * 32;     // [32][64]
  int nsn = L >> 6;
  int blk = blockIdx.x; int b = blk / nsn; int ns = blk - b * nsn;
  int t = threadIdx.x, lane = t & 31, mt = t >> 5;
  int m = lane & 15, kh = lane >> 4;
  int nthr = blockDim.x;
  v8f acc[4] = {};
  for (int kb = 0; kb < K; kb += 32) {
    for (int i = t; i < M * 2; i += nthr) {      // A: 32B per thread-step
      int row = i >> 1, half = i & 1;
      __builtin_memcpy(shA + row * 32 + half * 16,
                       W + (size_t)row * K + kb + half * 16, 32);
    }
    for (int i = t; i < 256; i += nthr) {        // B: 16B per thread-step
      int row = i >> 3, c8 = i & 7;
      __builtin_memcpy(shB + row * 64 + c8 * 8,
                       X + ((size_t)b * K + kb + row) * (size_t)L + ns * 64 + c8 * 8, 16);
    }
    if (kb + 32 < K && t < 32)
      __builtin_prefetch(X + ((size_t)b * K + kb + 32 + t) * (size_t)L + ns * 64, 0, 1);
    __syncthreads();
    v16bf a = frag_a(shA + (size_t)(mt * 16 + m) * 32, kh);
#pragma unroll
    for (int n = 0; n < 4; ++n) {
      v16bf bb = frag_b(shB + n * 16 + m, kh);
      acc[n] = wmma_bf16(a, bb, acc[n]);
    }
    __syncthreads();
  }
#pragma unroll
  for (int n = 0; n < 4; ++n) {
#pragma unroll
    for (int r8 = 0; r8 < 8; ++r8) {
      int o = mt * 16 + r8 + kh * 8;
      int l = ns * 64 + n * 16 + m;
      float v = acc[n][r8] + (bias ? bias[o] : 0.f);
      size_t oi = ((size_t)b * M + o) * (size_t)L + l;
      if (Yf)  Yf[oi]  = v;
      if (Ybf) Ybf[oi] = f2bf(v);
    }
  }
}

// ---------------------------------------------------------------------------
// Cooperative implicit-GEMM conv over taps. Block = one (batch, output row y),
// 12 waves = all 12 M-tiles (M=192). Zero-halo padded bf16 input [K,Hp,Hp];
// tap-major weights [ks*ks][192][K]. LDS: A 12KB + B 4KB.
// ---------------------------------------------------------------------------
DEV void conv_tap_block(const unsigned short* __restrict__ xpad,
                        const unsigned short* __restrict__ wt,
                        const float* __restrict__ bias,
                        int K, int ks, int hp, int y,
                        unsigned short* __restrict__ sh,
                        float* __restrict__ yf,
                        unsigned short* __restrict__ ybf,
                        unsigned short* __restrict__ ypad, int ypad_cs) {
  const int M = 192;
  unsigned short* shA = sh;                 // [192][32]
  unsigned short* shB = sh + 192 * 32;      // [32][64]
  const int Hp = 64 + 2 * hp, pad = ks >> 1;
  int t = threadIdx.x, lane = t & 31, mt = t >> 5;
  int m = lane & 15, kh = lane >> 4;
  v8f acc[4] = {};
  const int ntap = ks * ks;
  for (int tap = 0; tap < ntap; ++tap) {
    int r = tap / ks, s = tap - r * ks;
    int dy = r - pad, dx = s - pad;
    const unsigned short* wtap = wt + (size_t)tap * M * K;
    const unsigned short* xrow = xpad + (size_t)(y + hp + dy) * Hp + (hp + dx);
    for (int kb = 0; kb < K; kb += 32) {
      for (int i = t; i < M * 2; i += 384) {
        int row = i >> 1, half = i & 1;
        __builtin_memcpy(shA + row * 32 + half * 16,
                         wtap + (size_t)row * K + kb + half * 16, 32);
      }
      for (int i = t; i < 256; i += 384) {
        int row = i >> 3, c8 = i & 7;
        __builtin_memcpy(shB + row * 64 + c8 * 8,
                         xrow + (size_t)(kb + row) * Hp * Hp + c8 * 8, 16);
      }
      if (kb + 32 < K && t < 32)
        __builtin_prefetch(xrow + (size_t)(kb + 32 + t) * Hp * Hp, 0, 1);
      __syncthreads();
      v16bf a = frag_a(shA + (size_t)(mt * 16 + m) * 32, kh);
#pragma unroll
      for (int n = 0; n < 4; ++n) {
        v16bf bb = frag_b(shB + n * 16 + m, kh);
        acc[n] = wmma_bf16(a, bb, acc[n]);
      }
      __syncthreads();
    }
  }
#pragma unroll
  for (int n = 0; n < 4; ++n) {
#pragma unroll
    for (int r8 = 0; r8 < 8; ++r8) {
      int o = mt * 16 + r8 + kh * 8;
      int x = n * 16 + m;
      float v = acc[n][r8] + (bias ? bias[o] : 0.f);
      size_t lidx = (size_t)o * 4096 + (size_t)y * 64 + x;
      if (yf)  yf[lidx]  = v;
      if (ybf) ybf[lidx] = f2bf(v);
      if (ypad) ypad[(size_t)o * ypad_cs + (size_t)(y + 1) * 66 + (x + 1)] = f2bf(v);
    }
  }
}

// ---------------------------------------------------------------------------
// Gate: m = mean_{c,h} fmap1[0,c,h,w]; 12 adaptive bins; softmax; top-4 ids.
// ---------------------------------------------------------------------------
__global__ __launch_bounds__(256) void k_gate(const float* __restrict__ fmap1,
                                              int* __restrict__ idx) {
  __shared__ float part[256];
  __shared__ float m[64];
  __shared__ float binv[12];
  int t = threadIdx.x, w = t & 63, g = t >> 6;
  float acc = 0.f;
  for (int c = g; c < 192; c += 4) {
    const float* row = fmap1 + (size_t)c * 4096 + w;
    for (int hh = 0; hh < 64; ++hh) acc += row[hh * 64];
  }
  part[t] = acc; __syncthreads();
  if (t < 64) m[t] = (part[t] + part[64 + t] + part[128 + t] + part[192 + t]) * (1.f / 12288.f);
  __syncthreads();
  if (t < 12) {
    int lo = (t * 64) / 12, hi = ((t + 1) * 64 + 11) / 12;
    float s = 0.f; for (int i = lo; i < hi; ++i) s += m[i];
    binv[t] = s / (float)(hi - lo);
  }
  __syncthreads();
  if (t == 0) {
    float mx = binv[0]; for (int j = 1; j < 12; ++j) mx = fmaxf(mx, binv[j]);
    float p[12]; float s = 0.f;
    for (int j = 0; j < 12; ++j) { p[j] = __expf(binv[j] - mx); s += p[j]; }
    for (int j = 0; j < 12; ++j) p[j] /= s;
    unsigned used = 0;
    for (int k = 0; k < 4; ++k) {
      int best = 0; float bv = -1e30f;
      for (int j = 0; j < 12; ++j)
        if (!((used >> j) & 1) && p[j] > bv) { bv = p[j]; best = j; }
      used |= 1u << best; idx[k] = best;
    }
  }
}

// ---------------------------------------------------------------------------
// Expert slot (runtime dispatch on idx[slot]). Dense -> cooperative WMMA conv
// into padded concat (+bias); depthwise -> VALU. 384 threads / block.
// ---------------------------------------------------------------------------
struct ExPtrs { const float* w[12]; const float* b[12]; };

DEV size_t dense_woff(int e) {
  switch (e) {
    case 0: return 0u;       case 1: return 36864u;   case 2: return 368640u;
    case 3: return 1290240u; case 4: return 1327104u; default: return 1658880u;
  }
}

__global__ __launch_bounds__(384) void k_expert_slot(const int* __restrict__ idxp, int slot,
                                                     const float* __restrict__ I,
                                                     const unsigned short* __restrict__ xpad,
                                                     const unsigned short* __restrict__ wdense,
                                                     ExPtrs ep,
                                                     unsigned short* __restrict__ concat) {
  extern __shared__ unsigned short sh[];
  int e = idxp[slot];
  int ks = ((e % 3) == 0) ? 1 : (((e % 3) == 1) ? 3 : 5);
  if (e < 6) {  // dense conv, groups=1 -> cooperative WMMA implicit GEMM
    int blk = blockIdx.x; int b = blk >> 6; int y = blk & 63;
    conv_tap_block(xpad + (size_t)b * 192 * 68 * 68, wdense + dense_woff(e), ep.b[e],
                   192, ks, 2, y, sh, nullptr, nullptr,
                   concat + ((size_t)b * 768 + slot * 192) * (66 * 66), 66 * 66);
  } else {      // depthwise conv -> VALU
    const float* wd = ep.w[e]; const float* bb = ep.b[e];
    int pad = ks >> 1;
    size_t total = (size_t)4 * 192 * 4096;
    for (size_t i = (size_t)blockIdx.x * blockDim.x + threadIdx.x; i < total;
         i += (size_t)gridDim.x * blockDim.x) {
      int l = (int)(i & 4095); size_t bc = i >> 12;
      int b = (int)(bc / 192); int o = (int)(bc - (size_t)b * 192);
      int y = l >> 6, x = l & 63;
      const float* xr = I + (bc << 12);
      float acc = bb[o];
      for (int r = 0; r < ks; ++r) {
        int iy = y + r - pad; if (iy < 0 || iy > 63) continue;
        for (int s = 0; s < ks; ++s) {
          int ix = x + s - pad; if (ix < 0 || ix > 63) continue;
          acc += wd[(o * ks + r) * ks + s] * xr[iy * 64 + ix];
        }
      }
      concat[(((size_t)b * 768 + slot * 192 + o) * 66 + (y + 1)) * 66 + (x + 1)] = f2bf(acc);
    }
  }
}

// ex_out 3x3 conv: K=768 from padded concat -> fmap2 (f32) + fmap2_bf (bf16)
__global__ __launch_bounds__(384) void k_exout(const unsigned short* __restrict__ ypadin,
                                               const unsigned short* __restrict__ wt,
                                               const float* __restrict__ bias,
                                               float* __restrict__ yf,
                                               unsigned short* __restrict__ ybf) {
  extern __shared__ unsigned short sh[];
  int blk = blockIdx.x; int b = blk >> 6; int y = blk & 63;
  conv_tap_block(ypadin + (size_t)b * 768 * 66 * 66, wt, bias, 768, 3, 1, y, sh,
                 yf + (size_t)b * 192 * 4096, ybf + (size_t)b * 192 * 4096, nullptr, 0);
}

// depthwise 3x3, pad 1, no bias (kv_dw and q_dw), f32
__global__ void k_depthwise3(const float* __restrict__ X, const float* __restrict__ W,
                             float* __restrict__ Y, int Cn) {
  size_t total = (size_t)4 * Cn * 4096;
  size_t i = (size_t)blockIdx.x * blockDim.x + threadIdx.x;
  size_t st = (size_t)gridDim.x * blockDim.x;
  for (; i < total; i += st) {
    int l = (int)(i & 4095); size_t bc = i >> 12;
    int cc = (int)(bc % (size_t)Cn);
    int y = l >> 6, x = l & 63;
    const float* xr = X + (bc << 12);
    const float* wv = W + (size_t)cc * 9;
    float acc = 0.f;
#pragma unroll
    for (int r = 0; r < 3; ++r) {
      int iy = y + r - 1; if (iy < 0 || iy > 63) continue;
#pragma unroll
      for (int s = 0; s < 3; ++s) {
        int ix = x + s - 1; if (ix < 0 || ix > 63) continue;
        acc += wv[r * 3 + s] * xr[iy * 64 + ix];
      }
    }
    Y[i] = acc;
  }
}

// ---------------------------------------------------------------------------
// Final channel attention: S = (qn kn^T)*temp, softmax(32), out = S v -> bf16
// ---------------------------------------------------------------------------
__global__ __launch_bounds__(256) void k_attn2(const float* __restrict__ q,
                                               const float* __restrict__ kvd,
                                               const float* __restrict__ qinv,
                                               const float* __restrict__ kinv,
                                               const float* __restrict__ temp,
                                               unsigned short* __restrict__ aout) {
  __shared__ float qt[32][64];
  __shared__ float kt[32][64];
  __shared__ float S[32][32];
  __shared__ float vt[32][256];
  int bh = blockIdx.x; int b = bh / 6, h = bh - b * 6;
  int t = threadIdx.x;
  const float* qb = q   + ((size_t)b * 192 + h * 32) * 4096;
  const float* kb = kvd + ((size_t)b * 384 + h * 32) * 4096;
  const float* vb = kvd + ((size_t)b * 384 + 192 + h * 32) * 4096;
  int c = t >> 3, dg = t & 7;
  float g0 = 0, g1 = 0, g2 = 0, g3 = 0;
  for (int l0 = 0; l0 < 4096; l0 += 64) {
    for (int i = t; i < 32 * 64; i += 256) {
      int cc = i >> 6, ll = i & 63;
      qt[cc][ll] = qb[(size_t)cc * 4096 + l0 + ll];
      kt[cc][ll] = kb[(size_t)cc * 4096 + l0 + ll];
    }
    __syncthreads();
#pragma unroll 4
    for (int j = 0; j < 64; ++j) {
      float qv = qt[c][j];
      g0 += qv * kt[dg * 4 + 0][j]; g1 += qv * kt[dg * 4 + 1][j];
      g2 += qv * kt[dg * 4 + 2][j]; g3 += qv * kt[dg * 4 + 3][j];
    }
    __syncthreads();
  }
  float qi = qinv[b * 192 + h * 32 + c] * temp[h];
  S[c][dg * 4 + 0] = g0 * qi * kinv[b * 192 + h * 32 + dg * 4 + 0];
  S[c][dg * 4 + 1] = g1 * qi * kinv[b * 192 + h * 32 + dg * 4 + 1];
  S[c][dg * 4 + 2] = g2 * qi * kinv[b * 192 + h * 32 + dg * 4 + 2];
  S[c][dg * 4 + 3] = g3 * qi * kinv[b * 192 + h * 32 + dg * 4 + 3];
  __syncthreads();
  if (t < 32) {
    float mx = S[t][0];
    for (int d = 1; d < 32; ++d) mx = fmaxf(mx, S[t][d]);
    float sum = 0.f;
    for (int d = 0; d < 32; ++d) { float e = __expf(S[t][d] - mx); S[t][d] = e; sum += e; }
    float is = 1.f / sum;
    for (int d = 0; d < 32; ++d) S[t][d] *= is;
  }
  __syncthreads();
  for (int l0 = 0; l0 < 4096; l0 += 256) {
    for (int i = t; i < 32 * 256; i += 256) { int d = i >> 8, ll = i & 255; vt[d][ll] = vb[(size_t)d * 4096 + l0 + ll]; }
    __syncthreads();
    for (int cc = 0; cc < 32; ++cc) {
      float a0 = 0.f;
#pragma unroll 8
      for (int d = 0; d < 32; ++d) a0 += S[cc][d] * vt[d][t];
      aout[((size_t)b * 192 + h * 32 + cc) * 4096 + l0 + t] = f2bf(a0);
    }
    __syncthreads();
  }
}

// ---------------------------------------------------------------------------
// Host-side orchestration (graph-capture safe: only kernel launches on stream)
// ---------------------------------------------------------------------------
extern "C" void kernel_launch(void* const* d_in, const int* in_sizes, int n_in,
                              void* d_out, int out_size, void* d_ws, size_t ws_size,
                              hipStream_t stream) {
  (void)in_sizes; (void)n_in; (void)out_size; (void)ws_size;
  const float* I    = (const float*)d_in[0];
  const float* T    = (const float*)d_in[1];
  const float* temp = (const float*)d_in[2];
  const float* ca1w = (const float*)d_in[3];
  const float* exow = (const float*)d_in[4];
  const float* exob = (const float*)d_in[5];
  const float* kvw  = (const float*)d_in[6];
  const float* kvdw = (const float*)d_in[7];
  const float* qdw  = (const float*)d_in[8];
  const float* projw= (const float*)d_in[9];
  ExPtrs ep;
  for (int j = 0; j < 12; ++j) {
    ep.w[j] = (const float*)d_in[10 + 2 * j];
    ep.b[j] = (const float*)d_in[11 + 2 * j];
  }
  float* out = (float*)d_out;

  // workspace carve-out (~136 MB), 256B aligned, deterministic
  char* ws = (char*)d_ws; size_t off = 0;
  auto alloc = [&](size_t bytes) -> char* {
    char* p = ws + off; off = (off + bytes + 255) & ~(size_t)255; return p;
  };
  unsigned short* xpad   = (unsigned short*)alloc((size_t)4 * 192 * 68 * 68 * 2);
  unsigned short* concat = (unsigned short*)alloc((size_t)4 * 768 * 66 * 66 * 2);
  unsigned short* kn     = (unsigned short*)alloc((size_t)48 * 4096 * 2);
  float* qinv1           = (float*)alloc(192 * 4);
  unsigned short* outca  = (unsigned short*)alloc((size_t)192 * 4096 * 2);
  float* fmap1           = (float*)alloc((size_t)192 * 4096 * 4);
  int* idx               = (int*)alloc(64);
  unsigned short* wca1b  = (unsigned short*)alloc((size_t)192 * 192 * 2);
  unsigned short* wkvb   = (unsigned short*)alloc((size_t)384 * 192 * 2);
  unsigned short* wprojb = (unsigned short*)alloc((size_t)192 * 192 * 2);
  unsigned short* wexp   = (unsigned short*)alloc((size_t)2580480 * 2);
  unsigned short* wexout = (unsigned short*)alloc((size_t)1327104 * 2);
  float* fmap2           = (float*)alloc((size_t)4 * 192 * 4096 * 4);
  unsigned short* fmap2b = (unsigned short*)alloc((size_t)4 * 192 * 4096 * 2);
  float* kvbuf           = (float*)alloc((size_t)4 * 384 * 4096 * 4);
  float* kvd             = (float*)alloc((size_t)4 * 384 * 4096 * 4);
  float* qbuf            = (float*)alloc((size_t)4 * 192 * 4096 * 4);
  float* qinv2           = (float*)alloc(768 * 4);
  float* kinv2           = (float*)alloc(768 * 4);
  unsigned short* aout   = (unsigned short*)alloc((size_t)4 * 192 * 4096 * 2);

  auto cdiv = [](size_t a, size_t b) { return (int)((a + b - 1) / b); };
  const int smemConv = (192 * 32 + 32 * 64) * 2;          // 16 KB
  auto smemGemm = [](int M) { return (M * 32 + 32 * 64) * 2; };

  // 0) zero padded buffers (halo correctness), convert/repack weights to bf16
  k_zero32<<<1024, 256, 0, stream>>>((uint32_t*)xpad,   (size_t)4 * 192 * 68 * 68 / 2);
  k_zero32<<<2048, 256, 0, stream>>>((uint32_t*)concat, (size_t)4 * 768 * 66 * 66 / 2);
  k_f2bf<<<144, 256, 0, stream>>>(ca1w,  wca1b,  192 * 192);
  k_f2bf<<<288, 256, 0, stream>>>(kvw,   wkvb,   384 * 192);
  k_f2bf<<<144, 256, 0, stream>>>(projw, wprojb, 192 * 192);
  static const int    eks[6]  = {1, 3, 5, 1, 3, 5};
  static const size_t doff[6] = {0, 36864, 368640, 1290240, 1327104, 1658880};
  for (int j = 0; j < 6; ++j) {
    int n = 192 * 192 * eks[j] * eks[j];
    k_repack<<<cdiv(n, 256), 256, 0, stream>>>((const float*)d_in[10 + 2 * j],
                                               wexp + doff[j], 192, 192, eks[j]);
  }
  k_repack<<<cdiv(1327104, 256), 256, 0, stream>>>(exow, wexout, 192, 768, 3);
  k_pad_in<<<cdiv((size_t)4 * 192 * 4096, 256), 256, 0, stream>>>(I, xpad);

  // 1) stage-1 channel attention (batch 0 only) + 1x1 proj (WMMA)
  k_ca_pool_norm<<<48, 256, 0, stream>>>(T, kn);
  k_row_inv_norm<<<192, 256, 0, stream>>>(I, qinv1, 192, 192, 0, 4096);
  k_ca_attn<<<6, 256, 0, stream>>>(I, kn, qinv1, outca);
  k_gemm_coop<<<64, 384, smemGemm(192), stream>>>(wca1b, outca, fmap1,
      (unsigned short*)nullptr, (const float*)nullptr, 192, 192, 4096, 1);

  // 2) gate + top-4 experts (WMMA dense / VALU depthwise) + ex_out conv (WMMA)
  k_gate<<<1, 256, 0, stream>>>(fmap1, idx);
  for (int s = 0; s < 4; ++s)
    k_expert_slot<<<256, 384, smemConv, stream>>>(idx, s, I, xpad, wexp, ep, concat);
  k_exout<<<256, 384, smemConv, stream>>>(concat, wexout, exob, fmap2, fmap2b);

  // 3) kv 1x1 (WMMA, 24 waves/block) -> depthwise 3x3; q depthwise; l2 norms
  k_gemm_coop<<<256, 768, smemGemm(384), stream>>>(wkvb, fmap2b, kvbuf,
      (unsigned short*)nullptr, (const float*)nullptr, 384, 192, 4096, 4);
  k_depthwise3<<<4096, 256, 0, stream>>>(kvbuf, kvdw, kvd, 384);
  k_depthwise3<<<4096, 256, 0, stream>>>(fmap2, qdw, qbuf, 192);
  k_row_inv_norm<<<768, 256, 0, stream>>>(qbuf, qinv2, 192, 192, 0, 4096);
  k_row_inv_norm<<<768, 256, 0, stream>>>(kvd,  kinv2, 192, 384, 0, 4096);

  // 4) final channel attention + output 1x1 proj (WMMA) -> d_out (f32)
  k_attn2<<<24, 256, 0, stream>>>(qbuf, kvd, qinv2, kinv2, temp, aout);
  k_gemm_coop<<<256, 384, smemGemm(192), stream>>>(wprojb, aout, out,
      (unsigned short*)nullptr, (const float*)nullptr, 192, 192, 4096, 4);
}
